// DANet_17480516895193
// MI455X (gfx1250) — matrix operand order repaired and checked by reading, not compiled, not run on hardware
//
#include <hip/hip_runtime.h>

// Problem constants (from reference): B=4, C=64, H=W=64 -> N=4096, O=16.
#define BB 4
#define CC 64
#define NN 4096
#define OO 16
#define DP 32   // padded head dim for Q/K so one 16x16x32 WMMA covers the contraction

typedef __attribute__((ext_vector_type(16))) __bf16 v16bf;
typedef __attribute__((ext_vector_type(8)))  __bf16 v8bf;
typedef __attribute__((ext_vector_type(8)))  float  v8f;

union ABPack { v16bf v; v8bf h[2]; };

__device__ __forceinline__ v8f wmma_bf16(v16bf a, v16bf b, v8f c) {
  // 8 args: (neg_a, A, neg_b, B, c_mod, C, reuse_a, reuse_b)
  return __builtin_amdgcn_wmma_f32_16x16x32_bf16(false, a, false, b, (short)0, c,
                                                 false, false);
}

__device__ __forceinline__ v8f vzero8() {
  v8f z = {0.f, 0.f, 0.f, 0.f, 0.f, 0.f, 0.f, 0.f};
  return z;
}

// Async copy of 16 bytes global -> LDS (gfx1250 GLOBAL_LOAD_ASYNC_TO_LDS_B128).
// ldsoff is the byte offset within the workgroup LDS allocation (low 32 bits of
// a generic pointer to __shared__ memory — the LDS aperture occupies [63:32]).
__device__ __forceinline__ void async_b128(unsigned ldsoff, const void* gaddr) {
  asm volatile("global_load_async_to_lds_b128 %0, %1, off"
               :: "v"(ldsoff), "v"(gaddr) : "memory");
}
__device__ __forceinline__ void wait_async0() {
  asm volatile("s_wait_asynccnt 0" ::: "memory");
}

// A-fragment: M=16 x K=32, source stored row-major (M rows, leading dim ld elems).
// Lane L: row = L%16; lanes 0-15 carry K 0..15, lanes 16-31 carry K 16..31.
__device__ __forceinline__ v16bf load_a_frag(const __bf16* src, int ld) {
  int lane = threadIdx.x & 31;
  int row  = lane & 15;
  int half = lane >> 4;
  const __bf16* r = src + (size_t)row * ld;
  ABPack f;
  f.h[0] = *(const v8bf*)(r + half * 16);
  f.h[1] = *(const v8bf*)(r + half * 16 + 8);
  return f.v;
}

// B-fragment: K=32 x N=16, loaded from B^T stored row-major (16 rows x 32 K, ld elems).
// Lane L: col = L%16; lanes 0-15 carry K 0..15, lanes 16-31 carry K 16..31.
__device__ __forceinline__ v16bf load_b_frag(const __bf16* srcT, int ld) {
  int lane = threadIdx.x & 31;
  int col  = lane & 15;
  int half = lane >> 4;
  return *(const v16bf*)(srcT + (size_t)col * ld + half * 16);
}

__device__ __forceinline__ float rowmax16(float x) {
  #pragma unroll
  for (int m = 8; m >= 1; m >>= 1) x = fmaxf(x, __shfl_xor(x, m, 16));
  return x;
}
__device__ __forceinline__ float rowsum16(float x) {
  #pragma unroll
  for (int m = 8; m >= 1; m >>= 1) x += __shfl_xor(x, m, 16);
  return x;
}

// ---------------------------------------------------------------------------
// Prep: f32 -> bf16, both C-major (xbf: B,C,N) and N-major (xT: B,N,C) copies.
// ---------------------------------------------------------------------------
__global__ void danet_prep_x(const float* __restrict__ x,
                             __bf16* __restrict__ xbf,
                             __bf16* __restrict__ xT) {
  int i = blockIdx.x * 256 + threadIdx.x;      // i in [0, B*C*N)
  int b = i >> 18;                             // C*N = 2^18
  int r = i & 262143;
  int c = r >> 12;                             // N = 2^12
  int n = r & 4095;
  float v = x[i];
  xbf[i] = (__bf16)v;
  xT[((size_t)b << 18) + ((size_t)n << 6) + c] = (__bf16)v;
}

__global__ void danet_prep_w(const float* __restrict__ wq, const float* __restrict__ wk,
                             const float* __restrict__ wv,
                             __bf16* __restrict__ wqb, __bf16* __restrict__ wkb,
                             __bf16* __restrict__ wvb) {
  for (int i = threadIdx.x; i < OO * CC; i += blockDim.x) {
    wqb[i] = (__bf16)wq[i];
    wkb[i] = (__bf16)wk[i];
  }
  for (int i = threadIdx.x; i < CC * CC; i += blockDim.x) {
    wvb[i] = (__bf16)wv[i];
  }
}

// ---------------------------------------------------------------------------
// Channel attention: energy = x xT (64x64, contraction over N), max-subtract
// softmax over rows, store bf16 attention matrix.
// ---------------------------------------------------------------------------
__global__ void __launch_bounds__(256) danet_cam(const __bf16* __restrict__ xbf,
                                                 __bf16* __restrict__ attc) {
  __shared__ float energy[64][64];
  int b    = blockIdx.x;
  int wid  = threadIdx.x >> 5;
  int lane = threadIdx.x & 31;
  int half = lane >> 4;
  int col  = lane & 15;
  const __bf16* xb = xbf + ((size_t)b << 18);

  // 16 tiles of 16x16; 8 waves x 2 tiles each.
  #pragma unroll
  for (int tt = 0; tt < 2; ++tt) {
    int t  = wid * 2 + tt;
    int ci = (t >> 2) * 16;
    int di = (t & 3) * 16;
    v8f acc = vzero8();
    for (int n0 = 0; n0 < NN; n0 += 32) {
      v16bf a  = load_a_frag(xb + (size_t)ci * NN + n0, NN);
      v16bf bf = load_b_frag(xb + (size_t)di * NN + n0, NN);
      acc = wmma_bf16(a, bf, acc);
    }
    #pragma unroll
    for (int g = 0; g < 8; ++g)
      energy[ci + g + 8 * half][di + col] = acc[g];
  }
  __syncthreads();

  int r = threadIdx.x;
  if (r < 64) {
    float mx = -3.0e38f;
    for (int d = 0; d < 64; ++d) mx = fmaxf(mx, energy[r][d]);
    float tm = -3.0e38f;
    for (int d = 0; d < 64; ++d) tm = fmaxf(tm, mx - energy[r][d]);
    float s = 0.f;
    for (int d = 0; d < 64; ++d) s += __expf((mx - energy[r][d]) - tm);
    float inv = 1.f / s;
    for (int d = 0; d < 64; ++d)
      attc[((size_t)b << 12) + (r << 6) + d] =
          (__bf16)(__expf((mx - energy[r][d]) - tm) * inv);
  }
}

// ---------------------------------------------------------------------------
// Projections: Q,K (B,N,32 bf16, zero-padded depth) and V (B,C,N bf16) + bias.
// ---------------------------------------------------------------------------
__global__ void __launch_bounds__(128) danet_proj(
    const __bf16* __restrict__ xT, const __bf16* __restrict__ wqb,
    const __bf16* __restrict__ wkb, const __bf16* __restrict__ wvb,
    const float* __restrict__ bq, const float* __restrict__ bk,
    const float* __restrict__ bv,
    __bf16* __restrict__ Qb, __bf16* __restrict__ Kb, __bf16* __restrict__ Vb) {
  int bid  = blockIdx.x;
  int b    = bid >> 6;
  int grp  = bid & 63;
  int wid  = threadIdx.x >> 5;
  int lane = threadIdx.x & 31;
  int half = lane >> 4;
  int col  = lane & 15;
  int n0   = (grp * 4 + wid) * 16;
  const __bf16* xr = xT + (((size_t)(b << 12)) + n0) * 64;

  // Q, K tiles: rows = 16 spatial positions, cols = 16 output channels.
  v8f aq = vzero8(), ak = vzero8();
  #pragma unroll
  for (int ck = 0; ck < 64; ck += 32) {
    v16bf a = load_a_frag(xr + ck, 64);
    aq = wmma_bf16(a, load_b_frag(wqb + ck, 64), aq);
    ak = wmma_bf16(a, load_b_frag(wkb + ck, 64), ak);
  }
  float biasq = bq[col], biask = bk[col];
  #pragma unroll
  for (int g = 0; g < 8; ++g) {
    int n = n0 + g + 8 * half;
    size_t base = ((size_t)((b << 12) + n)) << 5;  // * DP
    Qb[base + col]      = (__bf16)(aq[g] + biasq);
    Qb[base + 16 + col] = (__bf16)0.f;             // depth pad
    Kb[base + col]      = (__bf16)(ak[g] + biask);
    Kb[base + 16 + col] = (__bf16)0.f;
  }

  // V tiles: rows = channels (4 tiles of 16), cols = 16 spatial positions.
  #pragma unroll
  for (int j = 0; j < 4; ++j) {
    v8f av = vzero8();
    #pragma unroll
    for (int ck = 0; ck < 64; ck += 32) {
      av = wmma_bf16(load_a_frag(wvb + (size_t)(j * 16) * 64 + ck, 64),
                     load_b_frag(xr + ck, 64), av);
    }
    #pragma unroll
    for (int g = 0; g < 8; ++g) {
      int c = j * 16 + g + 8 * half;
      Vb[(((size_t)b * 64 + c) << 12) + n0 + col] = (__bf16)(av[g] + bv[c]);
    }
  }
}

// ---------------------------------------------------------------------------
// Flash spatial attention: per block, 4 waves own 4 consecutive 16-query
// tiles; each 32-key iteration the block cooperatively stages the K tile
// (32x32 bf16) and V tile (64x32 bf16) into LDS with async loads
// (GLOBAL_LOAD_ASYNC_TO_LDS_B128 / ASYNCcnt), then all waves run score WMMAs,
// online softmax, and PV WMMAs from LDS. out_s stored f32 as (B, C, N).
// ---------------------------------------------------------------------------
__global__ void __launch_bounds__(128) danet_flash(
    const __bf16* __restrict__ Qb, const __bf16* __restrict__ Kb,
    const __bf16* __restrict__ Vb, float* __restrict__ outs) {
  __shared__ __bf16 Ktile[32][DP];      // 32 keys x 32 depth   (2 KB)
  __shared__ __bf16 Vtile[64][32];      // 64 chans x 32 keys   (4 KB)
  __shared__ __bf16 probs[4][16][32];   // per-wave prob tiles  (4 KB)

  int tid  = threadIdx.x;
  int bid  = blockIdx.x;
  int b    = bid >> 6;
  int qt   = bid & 63;
  int wid  = tid >> 5;
  int lane = tid & 31;
  int half = lane >> 4;
  int col  = lane & 15;
  int q0   = qt * 64 + wid * 16;

  v16bf qf = load_a_frag(Qb + (((size_t)(b << 12)) + q0) * DP, DP);

  float mrow[8], lrow[8];
  v8f acc[4];
  #pragma unroll
  for (int g = 0; g < 8; ++g) { mrow[g] = -3.0e38f; lrow[g] = 0.f; }
  #pragma unroll
  for (int j = 0; j < 4; ++j) acc[j] = vzero8();

  const __bf16* Kbase = Kb + ((size_t)(b << 12)) * DP;
  const __bf16* Vbase = Vb + (((size_t)b * 64) << 12);

  // Per-thread async-copy segment assignments (16 B = 8 bf16 per transfer).
  int kr = tid >> 2, ks = (tid & 3) * 8;        // K: 128 segments
  unsigned kloff = (unsigned)(size_t)&Ktile[kr][ks];
  const __bf16* kget = Kbase + (size_t)kr * DP + ks;
  int vr0 = tid >> 1;                            // V: 256 segments, 2 per thread
  unsigned vloff[2];
  const __bf16* vget[2];
  #pragma unroll
  for (int u = 0; u < 2; ++u) {
    int seg = tid + u * 128;
    int vr = seg >> 2, vc = (seg & 3) * 8;
    vloff[u] = (unsigned)(size_t)&Vtile[vr][vc];
    vget[u]  = Vbase + (size_t)vr * NN + vc;
  }
  (void)vr0;

  for (int n0 = 0; n0 < NN; n0 += 32) {
    // --- cooperative async staging of K/V tiles into LDS ---
    async_b128(kloff, kget + (size_t)n0 * DP);
    async_b128(vloff[0], vget[0] + n0);
    async_b128(vloff[1], vget[1] + n0);
    // speculative prefetch of the next K tile -> global_prefetch_b8
    __builtin_prefetch(kget + (size_t)(n0 + 32) * DP, 0, 0);
    wait_async0();
    __syncthreads();

    v16bf k0 = load_b_frag(&Ktile[0][0], DP);
    v16bf k1 = load_b_frag(&Ktile[16][0], DP);
    v8f s0 = wmma_bf16(qf, k0, vzero8());
    v8f s1 = wmma_bf16(qf, k1, vzero8());

    float alpha[8];
    #pragma unroll
    for (int g = 0; g < 8; ++g) {
      float cm = rowmax16(fmaxf(s0[g], s1[g]));
      float mn = fmaxf(mrow[g], cm);
      alpha[g] = __expf(mrow[g] - mn);
      mrow[g]  = mn;
      float p0 = __expf(s0[g] - mn);
      float p1 = __expf(s1[g] - mn);
      s0[g] = p0; s1[g] = p1;
      lrow[g] = lrow[g] * alpha[g] + rowsum16(p0 + p1);
    }

    // Stage probs (16 queries x 32 keys) in LDS, per-wave region.
    #pragma unroll
    for (int g = 0; g < 8; ++g) {
      probs[wid][g + 8 * half][col]      = (__bf16)s0[g];
      probs[wid][g + 8 * half][16 + col] = (__bf16)s1[g];
    }
    asm volatile("s_wait_dscnt 0" ::: "memory");  // LDS store -> LDS read RAW

    v16bf pf = load_a_frag(&probs[wid][0][0], 32);
    #pragma unroll
    for (int j = 0; j < 4; ++j) {
      #pragma unroll
      for (int g = 0; g < 8; ++g) acc[j][g] *= alpha[g];
      v16bf vf = load_b_frag(&Vtile[j * 16][0], 32);
      acc[j] = wmma_bf16(pf, vf, acc[j]);
    }
    __syncthreads();  // all waves done with K/V tiles before re-staging
  }

  #pragma unroll
  for (int j = 0; j < 4; ++j) {
    #pragma unroll
    for (int g = 0; g < 8; ++g) {
      float inv = 1.f / lrow[g];
      size_t idx = (((size_t)b * 64 + j * 16 + col) << 12) + (q0 + g + 8 * half);
      outs[idx] = acc[j][g] * inv;
    }
  }
}

// ---------------------------------------------------------------------------
// Final: out_c = att_c @ x (WMMA over d=64) fused with
// out = gamma_ca*out_c + gamma_sa*out_s + 2*x.
// ---------------------------------------------------------------------------
__global__ void __launch_bounds__(128) danet_final(
    const __bf16* __restrict__ attc, const __bf16* __restrict__ xT,
    const float* __restrict__ outs, const float* __restrict__ x,
    const float* __restrict__ gca, const float* __restrict__ gsa,
    float* __restrict__ out) {
  int tile = blockIdx.x * 4 + (threadIdx.x >> 5);
  int lane = threadIdx.x & 31;
  int half = lane >> 4;
  int col  = lane & 15;
  int b    = tile >> 10;
  int rem  = tile & 1023;
  int c0   = (rem >> 8) * 16;
  int n0   = (rem & 255) * 16;

  v8f acc = vzero8();
  #pragma unroll
  for (int ck = 0; ck < 64; ck += 32) {
    acc = wmma_bf16(load_a_frag(attc + ((size_t)b << 12) + (size_t)c0 * 64 + ck, 64),
                    load_b_frag(xT + (((size_t)(b << 12)) + n0) * 64 + ck, 64), acc);
  }
  float gc = gca[0], gs = gsa[0];
  #pragma unroll
  for (int g = 0; g < 8; ++g) {
    size_t idx = (((size_t)b * 64 + c0 + g + 8 * half) << 12) + n0 + col;
    out[idx] = gc * acc[g] + gs * outs[idx] + 2.f * x[idx];
  }
}

// ---------------------------------------------------------------------------
extern "C" void kernel_launch(void* const* d_in, const int* in_sizes, int n_in,
                              void* d_out, int out_size, void* d_ws, size_t ws_size,
                              hipStream_t stream) {
  (void)in_sizes; (void)n_in; (void)out_size; (void)ws_size;
  const float* x   = (const float*)d_in[0];
  const float* wq  = (const float*)d_in[1];
  const float* bq  = (const float*)d_in[2];
  const float* wk  = (const float*)d_in[3];
  const float* bk  = (const float*)d_in[4];
  const float* wv  = (const float*)d_in[5];
  const float* bv  = (const float*)d_in[6];
  const float* gca = (const float*)d_in[7];
  const float* gsa = (const float*)d_in[8];
  float* out = (float*)d_out;

  char* w = (char*)d_ws;
  size_t off = 0;
  auto alloc = [&](size_t bytes) -> void* {
    void* p = (void*)(w + off);
    off += (bytes + 255) & ~(size_t)255;
    return p;
  };

  __bf16* xbf  = (__bf16*)alloc((size_t)BB * CC * NN * 2);  // (B,C,N) bf16
  __bf16* xT   = (__bf16*)alloc((size_t)BB * NN * CC * 2);  // (B,N,C) bf16
  __bf16* Qb   = (__bf16*)alloc((size_t)BB * NN * DP * 2);  // (B,N,32) bf16
  __bf16* Kb   = (__bf16*)alloc((size_t)BB * NN * DP * 2);  // (B,N,32) bf16
  __bf16* Vb   = (__bf16*)alloc((size_t)BB * CC * NN * 2);  // (B,C,N) bf16
  __bf16* attc = (__bf16*)alloc((size_t)BB * CC * CC * 2);  // (B,64,64) bf16
  float*  outs = (float*) alloc((size_t)BB * CC * NN * 4);  // (B,C,N) f32
  __bf16* wqb  = (__bf16*)alloc((size_t)OO * CC * 2);
  __bf16* wkb  = (__bf16*)alloc((size_t)OO * CC * 2);
  __bf16* wvb  = (__bf16*)alloc((size_t)CC * CC * 2);

  danet_prep_x<<<(BB * CC * NN) / 256, 256, 0, stream>>>(x, xbf, xT);
  danet_prep_w<<<1, 256, 0, stream>>>(wq, wk, wv, wqb, wkb, wvb);
  danet_cam<<<BB, 256, 0, stream>>>(xbf, attc);
  danet_proj<<<BB * 64, 128, 0, stream>>>(xT, wqb, wkb, wvb, bq, bk, bv, Qb, Kb, Vb);
  danet_flash<<<BB * 64, 128, 0, stream>>>(Qb, Kb, Vb, outs);
  danet_final<<<BB * 256, 128, 0, stream>>>(attc, xT, outs, x, gca, gsa, out);
}
